// GNN_68135361184078
// MI455X (gfx1250) — matrix operand order repaired and checked by reading, not compiled, not run on hardware
//
#include <hip/hip_runtime.h>

// ======================================================================
// GNN forward for MI455X (gfx1250, wave32, WMMA).
// GEMMs use v_wmma_f32_16x16x32_bf16 with 2-term bf16 split (AhBh+AhBl+AlBh)
// => ~1.5e-5 relative error with full fp32 exponent range (fp16 would
// overflow: hop magnitudes reach ~1e5 in layer 2).
// SPMM (A@x over 800k directed edges) uses global_atomic_add_f32; x fits in
// the 192MB L2 so gathers stay on-chip.
// ======================================================================

typedef __attribute__((ext_vector_type(16))) __bf16          v16bf;
typedef __attribute__((ext_vector_type(16))) unsigned short  v16us;
typedef __attribute__((ext_vector_type(8)))  float           v8f;

__device__ __forceinline__ unsigned short f2bf(float f) {   // RNE f32->bf16
  unsigned u = __float_as_uint(f);
  u += 0x7FFFu + ((u >> 16) & 1u);
  return (unsigned short)(u >> 16);
}
__device__ __forceinline__ float bf2f(unsigned short h) {
  return __uint_as_float(((unsigned)h) << 16);
}
__device__ __forceinline__ void atomAddF(float* p, float v) {
  (void)__hip_atomic_fetch_add(p, v, __ATOMIC_RELAXED, __HIP_MEMORY_SCOPE_AGENT);
}

// ---------------------------------------------------------------------
// Pack weights into B-fragment order with hi/lo bf16 split.
// Storage index: ((s*NT + t)*32 + lane)*16 + e
// B mapping (CDNA5 ISA): n = t*16 + (lane&15), k = s*32 + (lane>>4)*8 + e (e<8), s*32 + 16 + (lane>>4)*8 + (e-8) (e>=8)
// Optionally sums up to 3 source matrices (folds Wa0+Wa1+Wa2).
// ---------------------------------------------------------------------
__global__ void prep_wfrag_k(const float* __restrict__ W0,
                             const float* __restrict__ W1,
                             const float* __restrict__ W2,
                             unsigned short* __restrict__ hi,
                             unsigned short* __restrict__ lo,
                             int Ksteps, int NT, int ncols)
{
  int tid = blockIdx.x * blockDim.x + threadIdx.x;
  int total = Ksteps * NT * 512;
  if (tid >= total) return;
  int e    = tid & 15;
  int lane = (tid >> 4) & 31;
  int st   = tid >> 9;
  int t    = st % NT;
  int s    = st / NT;
  int k = s * 32 + (e < 8 ? (lane >> 4) * 8 + e : 16 + (lane >> 4) * 8 + (e - 8));
  int n = t * 16 + (lane & 15);
  float v = 0.f;
  if (n < ncols) {
    v = W0[(size_t)k * ncols + n];
    if (W1) v += W1[(size_t)k * ncols + n];
    if (W2) v += W2[(size_t)k * ncols + n];
  }
  unsigned short h = f2bf(v);
  hi[tid] = h;
  lo[tid] = f2bf(v - bf2f(h));
}

// ---------------------------------------------------------------------
// WMMA GEMM: each wave computes a 16(M) x (NT*16)(N) strip, K in steps of 32.
// A fragment per documented 16-bit 16x32 layout: lane L -> row m = L%16,
//   e<8 : k = s*32 + (L>>4)*8 + e ; e>=8 : k = s*32 + 16 + (L>>4)*8 + (e-8)
// MODE 0: C = acc, MODE 1: C += acc, MODE 2: edge store (n<8) + bias.
// ---------------------------------------------------------------------
template<int MODE, int NT>
__global__ void wmma_gemm_k(const float* __restrict__ A, int lda,
                            const unsigned short* __restrict__ Whi,
                            const unsigned short* __restrict__ Wlo,
                            float* __restrict__ C, int ldc,
                            int Ksteps, int Mtiles,
                            const float* __restrict__ bias)
{
  int wv = blockIdx.x * (blockDim.x >> 5) + (threadIdx.x >> 5);
  if (wv >= Mtiles) return;                  // wave-uniform exit, EXEC stays all-1
  int lane = threadIdx.x & 31;
  int hl   = lane >> 4;
  int arow = wv * 16 + (lane & 15);

  v8f acc[NT];
#pragma unroll
  for (int t = 0; t < NT; ++t) acc[t] = (v8f)0.f;

  for (int s = 0; s < Ksteps; ++s) {
    const float* ap = A + (size_t)arow * lda + s * 32 + hl * 8;
    float4 a0 = *(const float4*)(ap + 0);
    float4 a1 = *(const float4*)(ap + 4);
    float4 a2 = *(const float4*)(ap + 16);
    float4 a3 = *(const float4*)(ap + 20);
    float af[16] = {a0.x,a0.y,a0.z,a0.w, a1.x,a1.y,a1.z,a1.w,
                    a2.x,a2.y,a2.z,a2.w, a3.x,a3.y,a3.z,a3.w};
    v16us ahu, alu;
#pragma unroll
    for (int i = 0; i < 16; ++i) {
      unsigned short h = f2bf(af[i]);
      ahu[i] = h;
      alu[i] = f2bf(af[i] - bf2f(h));
    }
    v16bf ahi = __builtin_bit_cast(v16bf, ahu);
    v16bf alo = __builtin_bit_cast(v16bf, alu);

    const unsigned short* wb = Whi + ((size_t)(s * NT) * 32 + lane) * 16;
    const unsigned short* wl = Wlo + ((size_t)(s * NT) * 32 + lane) * 16;
#pragma unroll
    for (int t = 0; t < NT; ++t) {
      v16bf bhi = __builtin_bit_cast(v16bf, *(const v16us*)(wb + t * 512));
      v16bf blo = __builtin_bit_cast(v16bf, *(const v16us*)(wl + t * 512));
      acc[t] = __builtin_amdgcn_wmma_f32_16x16x32_bf16(false, ahi, false, bhi,
                                                       (short)0, acc[t], false, false);
      acc[t] = __builtin_amdgcn_wmma_f32_16x16x32_bf16(false, ahi, false, blo,
                                                       (short)0, acc[t], false, false);
      acc[t] = __builtin_amdgcn_wmma_f32_16x16x32_bf16(false, alo, false, bhi,
                                                       (short)0, acc[t], false, false);
    }
  }

  // C layout: VGPR j -> m = wv*16 + hl*8 + j ; n = t*16 + (lane&15)
  int n    = lane & 15;
  int mrow = wv * 16 + hl * 8;
  if (MODE == 2) {
    if (n < 8) {
#pragma unroll
      for (int j = 0; j < 8; ++j)
        C[(size_t)(mrow + j) * ldc + n] = acc[0][j] + bias[n];
    }
  } else {
#pragma unroll
    for (int t = 0; t < NT; ++t)
#pragma unroll
      for (int j = 0; j < 8; ++j) {
        size_t idx = (size_t)(mrow + j) * ldc + t * 16 + n;
        if (MODE == 1) C[idx] += acc[t][j];
        else           C[idx]  = acc[t][j];
      }
  }
}

// ---------------------------------------------------------------------
// SPMM via atomics: Y[r] += X[c], Y[c] += X[r] for every undirected edge.
// One thread per (edge, 4 columns); float4 gathers (hit L2), f32 atomics.
// ---------------------------------------------------------------------
__global__ void spmm_atomic_k(const int* __restrict__ ru, const int* __restrict__ cu,
                              const float* __restrict__ X, float* __restrict__ Y,
                              int F, long total)
{
  long tid = (long)blockIdx.x * blockDim.x + threadIdx.x;
  if (tid >= total) return;
  int f4 = F >> 2;
  int e  = (int)(tid / f4);
  int c  = (int)(tid % f4) * 4;
  int r  = ru[e], q = cu[e];
  float4 xr = *(const float4*)(X + (size_t)r * F + c);
  float4 xq = *(const float4*)(X + (size_t)q * F + c);
  float* yr = Y + (size_t)r * F + c;
  float* yq = Y + (size_t)q * F + c;
  atomAddF(yr + 0, xq.x); atomAddF(yr + 1, xq.y);
  atomAddF(yr + 2, xq.z); atomAddF(yr + 3, xq.w);
  atomAddF(yq + 0, xr.x); atomAddF(yq + 1, xr.y);
  atomAddF(yq + 2, xr.z); atomAddF(yq + 3, xr.w);
}

// ---------------------------------------------------------------------
// Epilogue: alpha (cols 0..127) -> BN(relu(.)), beta (cols 128..255) -> BN(.)
// In-place on [N,256]. Inference BN: g*(v-m)*rsqrt(var+1e-3)+b
// ---------------------------------------------------------------------
__global__ void bn_epilogue_k(float* __restrict__ X,
                              const float* __restrict__ gA, const float* __restrict__ bA,
                              const float* __restrict__ mA, const float* __restrict__ vA,
                              const float* __restrict__ gB, const float* __restrict__ bB,
                              const float* __restrict__ mB, const float* __restrict__ vB,
                              long total)
{
  long tid = (long)blockIdx.x * blockDim.x + threadIdx.x;
  if (tid >= total) return;
  int col = (int)(tid & 255);
  float v = X[tid];
  if (col < 128) {
    v = fmaxf(v, 0.f);
    v = gA[col] * (v - mA[col]) * rsqrtf(vA[col] + 1e-3f) + bA[col];
  } else {
    int c = col - 128;
    v = gB[c] * (v - mB[c]) * rsqrtf(vB[c] + 1e-3f) + bB[c];
  }
  X[tid] = v;
}

// out[e][0:8] = Z[ru[e]] + Z[cu[e]]
__global__ void edge_out_k(const int* __restrict__ ru, const int* __restrict__ cu,
                           const float* __restrict__ Z, float* __restrict__ out, int E)
{
  int e = blockIdx.x * blockDim.x + threadIdx.x;
  if (e >= E) return;
  int r = ru[e], q = cu[e];
  const float4* zr = (const float4*)(Z + (size_t)r * 8);
  const float4* zq = (const float4*)(Z + (size_t)q * 8);
  float4 a0 = zr[0], a1 = zr[1], b0 = zq[0], b1 = zq[1];
  float4 o0 = {a0.x + b0.x, a0.y + b0.y, a0.z + b0.z, a0.w + b0.w};
  float4 o1 = {a1.x + b1.x, a1.y + b1.y, a1.z + b1.z, a1.w + b1.w};
  float4* op = (float4*)(out + (size_t)e * 8);
  op[0] = o0; op[1] = o1;
}

// ======================================================================
// Host orchestration.
// Input order (insertion-order flatten of setup_inputs()):
//  [0]=x [1]=edge_row [2]=edge_col
//  layer l (base = 3 + 20*l): Wa[0..5], Wb[0..5],
//     bn_a{gamma,beta,mean,var}, bn_b{gamma,beta,mean,var}
//  [43]=edge.W (256x8) [44]=edge.b (8)
// ======================================================================
extern "C" void kernel_launch(void* const* d_in, const int* in_sizes, int n_in,
                              void* d_out, int out_size, void* d_ws, size_t ws_size,
                              hipStream_t stream)
{
  const float* x0 = (const float*)d_in[0];
  const int*   ru = (const int*)d_in[1];
  const int*   cu = (const int*)d_in[2];
  const int N = in_sizes[0] / 128;     // 50000 (divisible by 16)
  const int E = in_sizes[1];           // 400000

  // workspace carve-out
  char* ws = (char*)d_ws;
  size_t off = 0;
  auto carve = [&](size_t bytes) -> void* {
    void* p = ws + off;
    off += (bytes + 255) & ~(size_t)255;
    return p;
  };
  float* xA = (float*)carve((size_t)N * 256 * 4);
  float* xB = (float*)carve((size_t)N * 256 * 4);
  float* h0 = (float*)carve((size_t)N * 256 * 4);
  float* h1 = (float*)carve((size_t)N * 256 * 4);
  float* Z  = (float*)carve((size_t)N * 8 * 4);
  unsigned short* whi = (unsigned short*)carve((size_t)8 * 8 * 512 * 2);
  unsigned short* wlo = (unsigned short*)carve((size_t)8 * 8 * 512 * 2);
  (void)ws_size; (void)n_in;

  const int Mtiles = N / 16;
  const int gemmBlocks = (Mtiles + 3) / 4;        // 4 waves / block (128 thr)

  auto prep = [&](const float* W0, const float* W1, const float* W2,
                  int Ks, int NT, int ncols) {
    int total = Ks * NT * 512;
    prep_wfrag_k<<<(total + 255) / 256, 256, 0, stream>>>(W0, W1, W2, whi, wlo,
                                                          Ks, NT, ncols);
  };
  auto gemm_store = [&](const float* A, int lda, float* C, int Ks) {
    wmma_gemm_k<0, 8><<<gemmBlocks, 128, 0, stream>>>(A, lda, whi, wlo, C, 256,
                                                      Ks, Mtiles, nullptr);
  };
  auto gemm_acc = [&](const float* A, int lda, float* C, int Ks) {
    wmma_gemm_k<1, 8><<<gemmBlocks, 128, 0, stream>>>(A, lda, whi, wlo, C, 256,
                                                      Ks, Mtiles, nullptr);
  };
  auto spmm = [&](const float* X, float* Y, int F) {
    hipMemsetAsync(Y, 0, (size_t)N * F * 4, stream);
    long total = (long)E * (F / 4);
    spmm_atomic_k<<<(int)((total + 255) / 256), 256, 0, stream>>>(ru, cu, X, Y, F, total);
  };

  for (int l = 0; l < 2; ++l) {
    const int din = (l == 0) ? 128 : 256;
    const int Ks  = din / 32;
    const float* xin = (l == 0) ? x0 : xA;
    float* xout      = (l == 0) ? xA : xB;
    const int base = 3 + 20 * l;
    const float* Wa[6], *Wb[6];
    for (int i = 0; i < 6; ++i) {
      Wa[i] = (const float*)d_in[base + i];
      Wb[i] = (const float*)d_in[base + 6 + i];
    }
    const float* gA = (const float*)d_in[base + 12];
    const float* bA = (const float*)d_in[base + 13];
    const float* mA = (const float*)d_in[base + 14];
    const float* vA = (const float*)d_in[base + 15];
    const float* gB = (const float*)d_in[base + 16];
    const float* bB = (const float*)d_in[base + 17];
    const float* mB = (const float*)d_in[base + 18];
    const float* vB = (const float*)d_in[base + 19];

    // alpha/beta = x @ (W0+W1+W2)
    prep(Wa[0], Wa[1], Wa[2], Ks, 8, 128); gemm_store(xin, din, xout,        Ks);
    prep(Wb[0], Wb[1], Wb[2], Ks, 8, 128); gemm_store(xin, din, xout + 128,  Ks);
    // hop0 = A @ x
    spmm(xin, h0, din);
    prep(Wa[3], nullptr, nullptr, Ks, 8, 128); gemm_acc(h0, din, xout,       Ks);
    prep(Wb[3], nullptr, nullptr, Ks, 8, 128); gemm_acc(h0, din, xout + 128, Ks);
    // hop1 = A @ hop0
    spmm(h0, h1, din);
    prep(Wa[4], nullptr, nullptr, Ks, 8, 128); gemm_acc(h1, din, xout,       Ks);
    prep(Wb[4], nullptr, nullptr, Ks, 8, 128); gemm_acc(h1, din, xout + 128, Ks);
    // hop2 = A^2 @ hop1
    spmm(h1, h0, din);
    spmm(h0, h1, din);
    prep(Wa[5], nullptr, nullptr, Ks, 8, 128); gemm_acc(h1, din, xout,       Ks);
    prep(Wb[5], nullptr, nullptr, Ks, 8, 128); gemm_acc(h1, din, xout + 128, Ks);
    // BN(relu(alpha)) | BN(beta), in place
    long total = (long)N * 256;
    bn_epilogue_k<<<(int)((total + 255) / 256), 256, 0, stream>>>(
        xout, gA, bA, mA, vA, gB, bB, mB, vB, total);
  }

  // Edge dense: Z = x @ W_edge + b  (256 -> 8, one 16-wide N tile, cols 8..15 zero)
  const float* We = (const float*)d_in[43];
  const float* be = (const float*)d_in[44];
  prep(We, nullptr, nullptr, 8, 1, 8);
  wmma_gemm_k<2, 1><<<gemmBlocks, 128, 0, stream>>>(xB, 256, whi, wlo, Z, 8,
                                                    8, Mtiles, be);
  // out[e] = Z[ru[e]] + Z[cu[e]]
  edge_out_k<<<(E + 255) / 256, 256, 0, stream>>>(ru, cu, Z, (float*)d_out, E);
}
